// Head_26276609917310
// MI455X (gfx1250) — matrix operand order repaired and checked by reading, not compiled, not run on hardware
//
#include <hip/hip_runtime.h>

// Attention head with column-axis softmax (softmax over dim=1, per reference),
// bf16 WMMA (v_wmma_f32_16x16x32_bf16) for all GEMMs, fp32 softmax math.
//
// B=8, S=2048, C=384, D=64.  scale = 1/sqrt(64) = 0.125.

#define B_ 8
#define S_ 2048
#define C_ 384
#define D_ 64
#define SCALE_ 0.125f
#define NEG_ (-3.0e38f)

typedef __bf16 bf16;
typedef bf16  v16bf __attribute__((ext_vector_type(16)));
typedef bf16  v8bf  __attribute__((ext_vector_type(8)));
typedef float v8f   __attribute__((ext_vector_type(8)));

static __device__ __forceinline__ v8f wmma_bf16(v16bf a, v16bf b, v8f c) {
  // 8 args: (neg_a, A, neg_b, B, c_mod, C, reuse_a, reuse_b)
  return __builtin_amdgcn_wmma_f32_16x16x32_bf16(false, a, false, b, (short)0, c,
                                                 false, false);
}

// A-fragment (16x32 bf16, MxK): lane L holds row m=L&15; 16 contiguous k
// starting at (L>>4)*16.  `row` points at this lane's row
// at the 32-wide k-chunk base.
static __device__ __forceinline__ v16bf load_a_frag(const bf16* row, int lane) {
  int k0 = ((lane >> 4) & 1) * 16;
  v8bf c0 = *(const v8bf*)(row + k0);
  v8bf c1 = *(const v8bf*)(row + k0 + 8);
  v16bf a;
#pragma unroll
  for (int e = 0; e < 8; ++e) { a[e] = c0[e]; a[8 + e] = c1[e]; }
  return a;
}

// B-fragment (32x16 bf16, KxN): lane L holds col n=L&15, 16 contiguous k
// values starting at (L>>4)*16.  `col` points at this lane's 16 k values.
static __device__ __forceinline__ v16bf load_b_frag(const bf16* col) {
  v8bf c0 = *(const v8bf*)(col);
  v8bf c1 = *(const v8bf*)(col + 8);
  v16bf b;
#pragma unroll
  for (int e = 0; e < 8; ++e) { b[e] = c0[e]; b[8 + e] = c1[e]; }
  return b;
}

// ---------------------------------------------------------------------------
// Kernel 1: transpose + cvt weights: W fp32[C][D] -> Wt bf16[3*D][C]
// ---------------------------------------------------------------------------
__global__ void prep_w(const float* __restrict__ Wq, const float* __restrict__ Wk,
                       const float* __restrict__ Wv, bf16* __restrict__ wt) {
  int idx = blockIdx.x * 256 + threadIdx.x;
  if (idx >= 3 * C_ * D_) return;
  int w = idx / (C_ * D_);
  int r = idx % (C_ * D_);
  int c = r / D_;
  int d = r % D_;
  const float* W = (w == 0) ? Wq : ((w == 1) ? Wk : Wv);
  wt[(size_t)(w * D_ + d) * C_ + c] = (bf16)W[c * D_ + d];
}

// ---------------------------------------------------------------------------
// Kernel 2: QKV projection.  256 thr = 8 waves; wave owns 16 rows of x.
// Emits Q,K bf16 row-major [B*S][64] and V transposed bf16 [B][64][S].
// ---------------------------------------------------------------------------
__global__ void __launch_bounds__(256) qkv_proj(const float* __restrict__ x,
                                                const bf16* __restrict__ wt,
                                                bf16* __restrict__ qb,
                                                bf16* __restrict__ kb,
                                                bf16* __restrict__ vt) {
  int lane = threadIdx.x & 31;
  int wave = threadIdx.x >> 5;
  int m0   = blockIdx.x * 128 + wave * 16;      // global row of this wave's tile
  int n    = lane & 15;
  int lh   = (lane >> 4) & 1;

  const float* xrow = x + (size_t)(m0 + n) * C_;

  v8f acc[12];
#pragma unroll
  for (int t = 0; t < 12; ++t) acc[t] = (v8f){0, 0, 0, 0, 0, 0, 0, 0};

  for (int kc = 0; kc < 12; ++kc) {             // K = 384 = 12 * 32
    const float* p = xrow + kc * 32 + lh * 16;
    v16bf a;
#pragma unroll
    for (int e = 0; e < 8; ++e) { a[e] = (bf16)p[e]; a[8 + e] = (bf16)p[e + 8]; }
#pragma unroll
    for (int nt = 0; nt < 12; ++nt) {           // 12 n-tiles: q(4) k(4) v(4)
      const bf16* bp = wt + (size_t)(nt * 16 + n) * C_ + kc * 32 + lh * 16;
      acc[nt] = wmma_bf16(a, load_b_frag(bp), acc[nt]);
    }
  }

#pragma unroll
  for (int nt = 0; nt < 12; ++nt) {
    int w  = nt >> 2;
    int d0 = (nt & 3) * 16;
#pragma unroll
    for (int r = 0; r < 8; ++r) {
      int  grow = m0 + r + lh * 8;              // global row (m = r + lh*8)
      bf16 val  = (bf16)acc[nt][r];
      if (w == 0) {
        qb[(size_t)grow * D_ + d0 + n] = val;
      } else if (w == 1) {
        kb[(size_t)grow * D_ + d0 + n] = val;
      } else {
        int b = grow >> 11, sr = grow & (S_ - 1);
        vt[((size_t)b * D_ + d0 + n) * S_ + sr] = val;
      }
    }
  }
}

// ---------------------------------------------------------------------------
// Kernel 3: column softmax stats.  Wave owns one 16-key column tile j0 and
// streams query tiles i = j0..S-1 with an online (flash-style) max/sum per
// column.  colmax/colsum are fp32 [B][S].
// ---------------------------------------------------------------------------
__global__ void __launch_bounds__(256) col_stats(const bf16* __restrict__ qb,
                                                 const bf16* __restrict__ kb,
                                                 float* __restrict__ colmax,
                                                 float* __restrict__ colsum) {
  int lane = threadIdx.x & 31;
  int wave = threadIdx.x >> 5;
  int b    = blockIdx.x >> 4;
  int j0   = (blockIdx.x & 15) * 128 + wave * 16;
  int n    = lane & 15;
  int lh   = (lane >> 4) & 1;

  const bf16* qbase = qb + (size_t)b * S_ * D_;
  const bf16* kbase = kb + (size_t)b * S_ * D_;

  // B-fragments of K^T for this column tile (invariant over the query loop).
  const bf16* krow = kbase + (size_t)(j0 + n) * D_ + lh * 16;
  v16bf bk0 = load_b_frag(krow);
  v16bf bk1 = load_b_frag(krow + 32);

  float m_run = NEG_, l_run = 0.0f;
  int   j = j0 + n;

  for (int qi = j0; qi < S_; qi += 16) {
    const bf16* qrow = qbase + (size_t)(qi + n) * D_;
    v16bf a0 = load_a_frag(qrow, lane);
    v16bf a1 = load_a_frag(qrow + 32, lane);
    v8f s = (v8f){0, 0, 0, 0, 0, 0, 0, 0};
    s = wmma_bf16(a0, bk0, s);
    s = wmma_bf16(a1, bk1, s);

    float vals[8];
    float tmax = NEG_;
#pragma unroll
    for (int r = 0; r < 8; ++r) {
      int   i  = qi + r + lh * 8;
      float sv = (j <= i) ? s[r] * SCALE_ : NEG_;
      vals[r]  = sv;
      tmax     = fmaxf(tmax, sv);
    }
    float mnew = fmaxf(m_run, tmax);
    l_run *= __expf(m_run - mnew);              // l_run==0 when m_run sentinel
#pragma unroll
    for (int r = 0; r < 8; ++r) {
      int i = qi + r + lh * 8;
      if (j <= i) l_run += __expf(vals[r] - mnew);
    }
    m_run = mnew;
  }

  // Merge the two lane halves (rows 0-7 vs 8-15 of each tile live 16 lanes apart).
  float m_o = __shfl_xor(m_run, 16);
  float l_o = __shfl_xor(l_run, 16);
  float mm  = fmaxf(m_run, m_o);
  float ll  = l_run * __expf(m_run - mm) + l_o * __expf(m_o - mm);
  if (lane < 16) {
    colmax[(size_t)b * S_ + j0 + lane] = mm;
    colsum[(size_t)b * S_ + j0 + lane] = ll;
  }
}

// ---------------------------------------------------------------------------
// Kernel 4: output.  Wave owns a 16-query tile; recomputes score tiles,
// forms P' = exp(s*scale - colmax[j]) / colsum[j] (0 where j>i), re-stripes
// P' through LDS into A-fragment layout, accumulates O += P' @ V with WMMA.
// ---------------------------------------------------------------------------
__global__ void __launch_bounds__(256) attn_out(const bf16* __restrict__ qb,
                                                const bf16* __restrict__ kb,
                                                const bf16* __restrict__ vt,
                                                const float* __restrict__ colmax,
                                                const float* __restrict__ colsum,
                                                float* __restrict__ out) {
  __shared__ bf16 pbuf[8][16][40];              // per-wave 16x32 P tile, padded
  int lane = threadIdx.x & 31;
  int wave = threadIdx.x >> 5;
  int b    = blockIdx.x >> 4;
  int q0   = (blockIdx.x & 15) * 128 + wave * 16;   // query tile (within batch)
  int n    = lane & 15;
  int lh   = (lane >> 4) & 1;

  const bf16*  qbase = qb + (size_t)b * S_ * D_;
  const bf16*  kbase = kb + (size_t)b * S_ * D_;
  const bf16*  vbase = vt + (size_t)b * D_ * S_;
  const float* cm    = colmax + (size_t)b * S_;
  const float* cs    = colsum + (size_t)b * S_;

  // Q A-fragments (invariant over the key loop).
  const bf16* qrow = qbase + (size_t)(q0 + n) * D_;
  v16bf aq0 = load_a_frag(qrow, lane);
  v16bf aq1 = load_a_frag(qrow + 32, lane);

  v8f oacc[4];
#pragma unroll
  for (int t = 0; t < 4; ++t) oacc[t] = (v8f){0, 0, 0, 0, 0, 0, 0, 0};

  bf16(*P)[40] = pbuf[wave];
  int nchunks = (q0 >> 5) + 1;                  // 32-key chunks covering j<=q0+15

  for (int jc = 0; jc < nchunks; ++jc) {
    int jbase = jc * 32;
#pragma unroll
    for (int t = 0; t < 2; ++t) {
      int j0 = jbase + t * 16;
      const bf16* krow = kbase + (size_t)(j0 + n) * D_ + lh * 16;
      v8f s = (v8f){0, 0, 0, 0, 0, 0, 0, 0};
      s = wmma_bf16(aq0, load_b_frag(krow), s);
      s = wmma_bf16(aq1, load_b_frag(krow + 32), s);

      int   j  = j0 + n;
      float mj = cm[j];
      float rs = 1.0f / cs[j];
#pragma unroll
      for (int r = 0; r < 8; ++r) {
        int   i = q0 + r + lh * 8;
        float p = (j <= i) ? __expf(s[r] * SCALE_ - mj) * rs : 0.0f;
        P[r + lh * 8][t * 16 + n] = (bf16)p;    // D-layout -> row-major LDS
      }
    }
    // Same-wave LDS store->load dependence; compiler inserts s_wait_dscnt.
    v16bf ap = load_a_frag(&P[n][0], lane);     // P as A-fragment (16x32)
#pragma unroll
    for (int nt = 0; nt < 4; ++nt) {
      const bf16* vrow = vbase + (size_t)(nt * 16 + n) * S_ + jbase + lh * 16;
      oacc[nt] = wmma_bf16(ap, load_b_frag(vrow), oacc[nt]);
    }
  }

  float* orow = out + ((size_t)b * S_ + q0) * D_;
#pragma unroll
  for (int nt = 0; nt < 4; ++nt)
#pragma unroll
    for (int r = 0; r < 8; ++r)
      orow[(size_t)(r + lh * 8) * D_ + nt * 16 + n] = oacc[nt][r];
}

// ---------------------------------------------------------------------------
extern "C" void kernel_launch(void* const* d_in, const int* in_sizes, int n_in,
                              void* d_out, int out_size, void* d_ws, size_t ws_size,
                              hipStream_t stream) {
  (void)in_sizes; (void)n_in; (void)out_size; (void)ws_size;
  const float* x  = (const float*)d_in[0];
  const float* Wq = (const float*)d_in[1];
  const float* Wk = (const float*)d_in[2];
  const float* Wv = (const float*)d_in[3];
  float* out = (float*)d_out;

  // Workspace layout (bytes), all 256-aligned:
  //   wt   : 3*64*384*2          = 147456
  //   qb   : 16384*64*2          = 2097152
  //   kb   : 2097152
  //   vt   : 2097152
  //   cmax : 8*2048*4            = 65536
  //   csum : 65536               -> total 6569984
  char* ws   = (char*)d_ws;
  bf16* wt   = (bf16*)(ws);
  bf16* qb   = (bf16*)(ws + 147456);
  bf16* kb   = (bf16*)(ws + 147456 + 2097152);
  bf16* vt   = (bf16*)(ws + 147456 + 2u * 2097152);
  float* cmx = (float*)(ws + 147456 + 3u * 2097152);
  float* csm = (float*)(ws + 147456 + 3u * 2097152 + 65536);

  prep_w   <<<(3 * C_ * D_ + 255) / 256, 256, 0, stream>>>(Wq, Wk, Wv, wt);
  qkv_proj <<<(B_ * S_) / 128, 256, 0, stream>>>(x, wt, qb, kb, vt);
  col_stats<<<(B_ * S_) / 128, 256, 0, stream>>>(qb, kb, cmx, csm);
  attn_out <<<(B_ * S_) / 128, 256, 0, stream>>>(qb, kb, vt, cmx, csm, out);
}
